// L4Attention_24850680775017
// MI455X (gfx1250) — matrix operand order repaired and checked by reading, not compiled, not run on hardware
//
#include <hip/hip_runtime.h>
#include <hip/hip_bf16.h>

// ---------------- problem constants ----------------
#define T_SEQ 2048
#define C_DIM 5120
#define NH    40
#define NKV   8
#define HD    128
#define QKD   (NH * HD)   // 5120
#define KVD   (NKV * HD)  // 1024

typedef __attribute__((ext_vector_type(16))) __bf16 v16bf;
typedef __attribute__((ext_vector_type(8)))  __bf16 v8bf;
typedef __attribute__((ext_vector_type(8)))  float  v8f;

// fp32 -> bf16, round-to-nearest-even
static __device__ __forceinline__ __bf16 f2bf(float f) {
  unsigned u = __builtin_bit_cast(unsigned, f);
  u += 0x7FFFu + ((u >> 16) & 1u);
  unsigned short h = (unsigned short)(u >> 16);
  return __builtin_bit_cast(__bf16, h);
}

static __device__ __forceinline__ v8f zero8() {
  v8f z;
#pragma unroll
  for (int i = 0; i < 8; ++i) z[i] = 0.f;
  return z;
}

static __device__ __forceinline__ v8f wmma_bf16(v16bf a, v16bf b, v8f c) {
  // D = A(16x32 bf16) x B(32x16 bf16) + C(16x16 f32)
  return __builtin_amdgcn_wmma_f32_16x16x32_bf16(
      /*neg_a=*/false, a, /*neg_b=*/false, b,
      /*c_mod=*/(short)0, c, /*reuse_a=*/false, /*reuse_b=*/false);
}

// A fragment (16x32, row-major source, leading dim `ld` in elements).
// Contiguous K order used here: lane L -> row (L&15); K-half (L>>4): elems 0..15 = K kh*16+0..15.
// Two 16-byte loads per lane.
static __device__ __forceinline__ v16bf load_a_frag(const __bf16* base, int ld) {
  int lane = threadIdx.x & 31;
  int m = lane & 15, kh = lane >> 4;
  const __bf16* p = base + (size_t)m * ld + kh * 16;
  v8bf lo = *(const v8bf*)p;
  v8bf hi = *(const v8bf*)(p + 8);
  v16bf a;
#pragma unroll
  for (int i = 0; i < 8; ++i) { a[i] = lo[i]; a[i + 8] = hi[i]; }
  return a;
}

// B fragment (32x16). Source is row-major [N,K] (weights W[n][k]); we compute
// A*B^T so B-column n == W row n. Lane L -> column (L&15), K-half (L>>4).
// One contiguous 32-byte load per lane.
static __device__ __forceinline__ v16bf load_b_frag(const __bf16* base, int ld) {
  int lane = threadIdx.x & 31;
  int n = lane & 15, kh = lane >> 4;
  return *(const v16bf*)(base + (size_t)n * ld + kh * 16);
}

// ---------------- elementwise kernels ----------------
__global__ void cvt_f32_to_bf16(const float* __restrict__ src,
                                __bf16* __restrict__ dst, size_t n) {
  size_t i = (size_t)blockIdx.x * blockDim.x + threadIdx.x;
  size_t stride = (size_t)gridDim.x * blockDim.x;
  for (; i < n; i += stride) dst[i] = f2bf(src[i]);
}

// src: f32 [T, heads, HD]; dst: bf16 [heads][T][HD], RoPE'd (interleaved pairs)
__global__ void rope_to_bf16(const float* __restrict__ src,
                             __bf16* __restrict__ dst, int heads) {
  size_t idx = (size_t)blockIdx.x * blockDim.x + threadIdx.x;
  size_t total = (size_t)T_SEQ * heads * (HD / 2);
  if (idx >= total) return;
  int pair = (int)(idx % (HD / 2));
  int h = (int)((idx / (HD / 2)) % heads);
  int t = (int)(idx / ((size_t)(HD / 2) * heads));
  size_t si = ((size_t)t * heads + h) * HD + 2 * pair;
  float a = src[si], b = src[si + 1];
  const float LOG_BASE = 13.122363377404328f;  // ln(500000)
  float invf = __expf(-((float)(2 * pair) / (float)HD) * LOG_BASE);
  float ang = (float)t * invf;
  float s, c;
  __sincosf(ang, &s, &c);
  size_t di = ((size_t)h * T_SEQ + t) * HD + 2 * pair;
  dst[di]     = f2bf(a * c - b * s);
  dst[di + 1] = f2bf(a * s + b * c);
}

// src: f32 [T, NKV*HD]; dst: bf16 [NKV][HD][T]  (V transposed for PV B-frags)
__global__ void transpose_v_bf16(const float* __restrict__ src,
                                 __bf16* __restrict__ dst) {
  size_t idx = (size_t)blockIdx.x * blockDim.x + threadIdx.x;
  size_t total = (size_t)T_SEQ * KVD;
  if (idx >= total) return;
  int d = (int)(idx % HD);
  int h = (int)((idx / HD) % NKV);
  int t = (int)(idx / KVD);
  dst[((size_t)h * HD + d) * T_SEQ + t] = f2bf(src[idx]);
}

// ---------------- WMMA GEMM: C[M,N] = A[M,K] * B[N,K]^T (f32 out) -----------
// block = 256 threads = 8 waves (2 x 4); wave tile 64x64 (4x4 WMMA tiles);
// block tile 128x256. 16 WMMAs vs 8 fragment loads per K-step (2x register
// reuse of A and B -> matrix-pipe bound, not vmem bound).
// M%128==0, N%256==0, K%32==0 hold for all calls here.
__global__ __launch_bounds__(256) void gemm_bf16_nt(
    const __bf16* __restrict__ A, const __bf16* __restrict__ B,
    float* __restrict__ C, int M, int N, int K) {
  int wave = threadIdx.x >> 5;
  int lane = threadIdx.x & 31;
  int wm = wave >> 2, wn = wave & 3;
  int m0 = blockIdx.x * 128 + wm * 64;
  int n0 = blockIdx.y * 256 + wn * 64;

  v8f acc[4][4];
#pragma unroll
  for (int i = 0; i < 4; ++i)
#pragma unroll
    for (int j = 0; j < 4; ++j) acc[i][j] = zero8();

  for (int k0 = 0; k0 < K; k0 += 32) {
    if (k0 + 32 < K) {
      __builtin_prefetch(A + (size_t)m0 * K + k0 + 32, 0, 3);
      __builtin_prefetch(B + (size_t)n0 * K + k0 + 32, 0, 3);
    }
    v16bf a[4], b[4];
#pragma unroll
    for (int i = 0; i < 4; ++i)
      a[i] = load_a_frag(A + (size_t)(m0 + 16 * i) * K + k0, K);
#pragma unroll
    for (int j = 0; j < 4; ++j)
      b[j] = load_b_frag(B + (size_t)(n0 + 16 * j) * K + k0, K);
#pragma unroll
    for (int i = 0; i < 4; ++i)
#pragma unroll
      for (int j = 0; j < 4; ++j) acc[i][j] = wmma_bf16(a[i], b[j], acc[i][j]);
  }

  // D layout: lane L holds rows (L>>4)*8 + r (r=0..7), column L&15.
  int g = lane >> 4, n = lane & 15;
#pragma unroll
  for (int im = 0; im < 4; ++im)
#pragma unroll
    for (int in = 0; in < 4; ++in)
#pragma unroll
      for (int r = 0; r < 8; ++r)
        C[(size_t)(m0 + im * 16 + g * 8 + r) * N + n0 + in * 16 + n] =
            acc[im][in][r];
}

// ---------------- flash attention (causal, GQA) ----------------
// One wave per (head, 16-query tile). Q: bf16 [NH][T][HD] (RoPE'd),
// K: bf16 [NKV][T][HD] (RoPE'd), Vt: bf16 [NKV][HD][T].
// Y out: bf16 [T][NH*HD].
__global__ __launch_bounds__(256) void attn_fwd(
    const __bf16* __restrict__ Q, const __bf16* __restrict__ Kh,
    const __bf16* __restrict__ Vt, __bf16* __restrict__ Y) {
  // per-wave P staging tile: 16 rows x 32 cols, row stride 40 (conflict pad)
  __shared__ __align__(32) __bf16 ldsP[8][16 * 40];

  int wave = threadIdx.x >> 5, lane = threadIdx.x & 31;
  int task = blockIdx.x * 8 + wave;          // NH * (T/16) = 5120 tasks
  int h = task >> 7;                         // / 128 q-tiles
  int q0 = (task & 127) << 4;
  int kvh = h / (NH / NKV);

  const __bf16* Qb = Q + ((size_t)h * T_SEQ + q0) * HD;
  const __bf16* Kb = Kh + (size_t)kvh * T_SEQ * HD;
  const __bf16* Vb = Vt + (size_t)kvh * HD * T_SEQ;

  v16bf qf[4];
#pragma unroll
  for (int dk = 0; dk < 4; ++dk) qf[dk] = load_a_frag(Qb + dk * 32, HD);

  v8f o[8];
#pragma unroll
  for (int j = 0; j < 8; ++j) o[j] = zero8();
  float rm[8], rl[8];
#pragma unroll
  for (int r = 0; r < 8; ++r) { rm[r] = -1e30f; rl[r] = 0.f; }

  int g = lane >> 4, n = lane & 15;
  const float scale = 0.08838834764831845f;  // 1/sqrt(128)
  __bf16* pp = &ldsP[wave][0];

  for (int kb = 0; kb < q0 + 16; kb += 32) {
    // S = Q * K^T for 32 keys (two 16-key halves), f32 accumulate
    v8f s[2];
    s[0] = zero8(); s[1] = zero8();
#pragma unroll
    for (int half = 0; half < 2; ++half) {
      const __bf16* kbase = Kb + (size_t)(kb + half * 16) * HD;
#pragma unroll
      for (int dk = 0; dk < 4; ++dk)
        s[half] = wmma_bf16(qf[dk], load_b_frag(kbase + dk * 32, HD), s[half]);
    }

    // scale + causal bias (-1e9 like the reference), block row max
    float mx[8];
#pragma unroll
    for (int r = 0; r < 8; ++r) {
      int row = q0 + g * 8 + r;
      float s0 = s[0][r] * scale + ((kb + n) > row ? -1e9f : 0.f);
      float s1 = s[1][r] * scale + ((kb + 16 + n) > row ? -1e9f : 0.f);
      s[0][r] = s0; s[1][r] = s1;
      mx[r] = fmaxf(s0, s1);
    }
#pragma unroll
    for (int off = 1; off < 16; off <<= 1)
#pragma unroll
      for (int r = 0; r < 8; ++r)
        mx[r] = fmaxf(mx[r], __shfl_xor(mx[r], off, 32));

    float al[8], ps[8];
#pragma unroll
    for (int r = 0; r < 8; ++r) {
      float nm = fmaxf(rm[r], mx[r]);
      al[r] = __expf(rm[r] - nm);
      rm[r] = nm;
      float p0 = __expf(s[0][r] - nm);
      float p1 = __expf(s[1][r] - nm);
      s[0][r] = p0; s[1][r] = p1;
      ps[r] = p0 + p1;
    }
#pragma unroll
    for (int off = 1; off < 16; off <<= 1)
#pragma unroll
      for (int r = 0; r < 8; ++r) ps[r] += __shfl_xor(ps[r], off, 32);
#pragma unroll
    for (int r = 0; r < 8; ++r) rl[r] = rl[r] * al[r] + ps[r];

    // rescale running output
#pragma unroll
    for (int j = 0; j < 8; ++j)
#pragma unroll
      for (int r = 0; r < 8; ++r) o[j][r] *= al[r];

    // stage P (16x32 bf16) through LDS; same-wave RAW -> compiler dscnt waits
#pragma unroll
    for (int half = 0; half < 2; ++half)
#pragma unroll
      for (int r = 0; r < 8; ++r)
        pp[(size_t)(g * 8 + r) * 40 + half * 16 + n] = f2bf(s[half][r]);

    v16bf pf = load_a_frag(pp, 40);

    // O += P * V  (B columns are contiguous thanks to [HKV][HD][T] layout)
#pragma unroll
    for (int j = 0; j < 8; ++j)
      o[j] = wmma_bf16(pf, load_b_frag(Vb + (size_t)(j * 16) * T_SEQ + kb,
                                       T_SEQ), o[j]);
  }

  // normalize and write Y[t][h*HD + d] as bf16
#pragma unroll
  for (int r = 0; r < 8; ++r) rl[r] = 1.f / rl[r];
#pragma unroll
  for (int j = 0; j < 8; ++j)
#pragma unroll
    for (int r = 0; r < 8; ++r)
      Y[(size_t)(q0 + g * 8 + r) * QKD + h * HD + j * 16 + n] =
          f2bf(o[j][r] * rl[r]);
}

// ---------------- host launcher ----------------
extern "C" void kernel_launch(void* const* d_in, const int* in_sizes, int n_in,
                              void* d_out, int out_size, void* d_ws,
                              size_t ws_size, hipStream_t stream) {
  (void)in_sizes; (void)n_in; (void)out_size; (void)ws_size;
  const float* x  = (const float*)d_in[0];
  const float* Wq = (const float*)d_in[1];
  const float* Wk = (const float*)d_in[2];
  const float* Wv = (const float*)d_in[3];
  const float* Wo = (const float*)d_in[4];
  // d_in[5] (attn_bias) is exactly the causal -1e9 mask; applied analytically.
  float* out = (float*)d_out;

  char* p = (char*)d_ws;
  auto take = [&](size_t bytes) -> char* {
    char* r = p;
    p += (bytes + 255) & ~(size_t)255;
    return r;
  };
  __bf16* xh  = (__bf16*)take((size_t)T_SEQ * C_DIM * 2);
  __bf16* WqH = (__bf16*)take((size_t)QKD * C_DIM * 2);
  __bf16* WkH = (__bf16*)take((size_t)KVD * C_DIM * 2);
  __bf16* WvH = (__bf16*)take((size_t)KVD * C_DIM * 2);
  __bf16* WoH = (__bf16*)take((size_t)C_DIM * QKD * 2);
  float*  Qf  = (float*)take((size_t)T_SEQ * QKD * 4);
  float*  Kf  = (float*)take((size_t)T_SEQ * KVD * 4);
  float*  Vf  = (float*)take((size_t)T_SEQ * KVD * 4);
  __bf16* QhB = (__bf16*)take((size_t)NH * T_SEQ * HD * 2);
  __bf16* KhB = (__bf16*)take((size_t)NKV * T_SEQ * HD * 2);
  __bf16* VtB = (__bf16*)take((size_t)NKV * HD * T_SEQ * 2);
  __bf16* Yh  = (__bf16*)take((size_t)T_SEQ * QKD * 2);

  // 1) fp32 -> bf16 conversions
  cvt_f32_to_bf16<<<4096, 256, 0, stream>>>(x,  xh,  (size_t)T_SEQ * C_DIM);
  cvt_f32_to_bf16<<<4096, 256, 0, stream>>>(Wq, WqH, (size_t)QKD * C_DIM);
  cvt_f32_to_bf16<<<4096, 256, 0, stream>>>(Wk, WkH, (size_t)KVD * C_DIM);
  cvt_f32_to_bf16<<<4096, 256, 0, stream>>>(Wv, WvH, (size_t)KVD * C_DIM);
  cvt_f32_to_bf16<<<4096, 256, 0, stream>>>(Wo, WoH, (size_t)C_DIM * QKD);

  // 2) QKV projections (WMMA)
  gemm_bf16_nt<<<dim3(T_SEQ / 128, QKD / 256), 256, 0, stream>>>(
      xh, WqH, Qf, T_SEQ, QKD, C_DIM);
  gemm_bf16_nt<<<dim3(T_SEQ / 128, KVD / 256), 256, 0, stream>>>(
      xh, WkH, Kf, T_SEQ, KVD, C_DIM);
  gemm_bf16_nt<<<dim3(T_SEQ / 128, KVD / 256), 256, 0, stream>>>(
      xh, WvH, Vf, T_SEQ, KVD, C_DIM);

  // 3) RoPE (+ head-major transpose) and V transpose
  {
    size_t tq = (size_t)T_SEQ * NH * (HD / 2);
    rope_to_bf16<<<(unsigned)((tq + 255) / 256), 256, 0, stream>>>(Qf, QhB, NH);
    size_t tk = (size_t)T_SEQ * NKV * (HD / 2);
    rope_to_bf16<<<(unsigned)((tk + 255) / 256), 256, 0, stream>>>(Kf, KhB, NKV);
    size_t tv = (size_t)T_SEQ * KVD;
    transpose_v_bf16<<<(unsigned)((tv + 255) / 256), 256, 0, stream>>>(Vf, VtB);
  }

  // 4) causal flash attention (WMMA), 5120 wave-tasks / 8 waves per block
  attn_fwd<<<(NH * (T_SEQ / 16)) / 8, 256, 0, stream>>>(QhB, KhB, VtB, Yh);

  // 5) output projection -> f32 d_out
  gemm_bf16_nt<<<dim3(T_SEQ / 128, C_DIM / 256), 256, 0, stream>>>(
      Yh, WoH, out, T_SEQ, C_DIM, QKD);
}
